// QNet_84129819394104
// MI455X (gfx1250) — matrix operand order repaired and checked by reading, not compiled, not run on hardware
//
#include <hip/hip_runtime.h>
#include <hip/hip_bf16.h>

typedef __attribute__((ext_vector_type(16))) _Float16 v16h;
typedef __attribute__((ext_vector_type(8)))  _Float16 v8h;
typedef __attribute__((ext_vector_type(8)))  float    v8f;

#define NREP 128

// Build a 16x32 f16 A-fragment from row-major LDS (row stride in halves given by caller).
// Lane layout (ISA 7.12.2): lanes<16 hold K {0..7,16..23}, lanes>=16 hold K {8..15,24..31}.
static __device__ __forceinline__ v16h ldsA(const _Float16* base) {
  v8h lo = *(const v8h*)(base);        // K = kbase .. kbase+7
  v8h hi = *(const v8h*)(base + 16);   // K = kbase+16 .. kbase+23
  v16h a;
#pragma unroll
  for (int e = 0; e < 8; ++e) { a[e] = lo[e]; a[e + 8] = hi[e]; }
  return a;
}

// B-fragment: 16 contiguous pre-swizzled f16 per lane in global scratch.
static __device__ __forceinline__ v16h ldgB(const _Float16* p) {
  v8h lo = *(const v8h*)(p);
  v8h hi = *(const v8h*)(p + 8);
  v16h b;
#pragma unroll
  for (int e = 0; e < 8; ++e) { b[e] = lo[e]; b[e + 8] = hi[e]; }
  return b;
}

// ---------------- weight prep: f32 -> f16 B-fragments ----------------
// W2 frags: idx = ((n*4+nt)*4+k)*512 + lane*16 + e, K = k*32 + (lane>=16?8:0) + (e<8?e:e+8), N = nt*16+(lane&15)
// Wc frags (top half K<64): idx = ((n*4+nt)*2+k)*512 + lane*16 + e
__global__ __launch_bounds__(256) void qnet_prep(const float* __restrict__ W2,
                                                 const float* __restrict__ Wc,
                                                 _Float16* __restrict__ w2h,
                                                 _Float16* __restrict__ wch) {
  int id = blockIdx.x * 256 + threadIdx.x;
  if (id < 65536) {
    int e = id & 15, l = (id >> 4) & 31, blk = id >> 9;
    int k = blk & 3, nt = (blk >> 2) & 3, n = blk >> 4;
    int K = k * 32 + ((l >= 16) ? 8 : 0) + ((e < 8) ? e : e + 8);
    int N = nt * 16 + (l & 15);
    w2h[id] = (_Float16)W2[n * 8192 + K * 64 + N];
  } else if (id < 98304) {
    int id2 = id - 65536;
    int e = id2 & 15, l = (id2 >> 4) & 31, blk = id2 >> 9;
    int k = blk & 1, nt = (blk >> 1) & 3, n = blk >> 3;
    int K = k * 32 + ((l >= 16) ? 8 : 0) + ((e < 8) ? e : e + 8);
    int N = nt * 16 + (l & 15);
    wch[id2] = (_Float16)Wc[n * 8192 + K * 64 + N];
  }
}

__global__ __launch_bounds__(256) void qnet_zero(float* __restrict__ rep) {
  int id = blockIdx.x * 256 + threadIdx.x;
  if (id < NREP * 512) rep[id] = 0.f;
}

// ---------------- phase 1: h + mask mixing + batch reduction ----------------
__global__ __launch_bounds__(256) void qnet_phase1(
    const float* __restrict__ obs, const float* __restrict__ W1,
    const float* __restrict__ b1, const _Float16* __restrict__ w2h,
    const float* __restrict__ b2, float* __restrict__ rep)
{
  __shared__ float    obsL[512];          // 32 batch x 8 agents x 2
  __shared__ _Float16 h1L[32 * 136];      // layer-1 activations (padded stride)
  __shared__ _Float16 hAll[32 * 520];     // h for all agents: [b][n*64+c]
  __shared__ unsigned mrowL[256];         // mask row bits per (b,i)
  __shared__ float    invJL[256];
  __shared__ float    commL[512];         // per-WG partial comm accumulator

  const int t    = threadIdx.x;
  const int lane = t & 31;
  const int w    = t >> 5;
  const int bl   = t >> 3;                // 0..31
  const int b0   = blockIdx.x * 32;

  obsL[t]       = obs[b0 * 16 + t];
  obsL[t + 256] = obs[b0 * 16 + 256 + t];
  commL[t] = 0.f; commL[t + 256] = 0.f;
  __syncthreads();

  const int mt = w & 1, nt = w >> 1;          // 8 waves -> 2x4 output tiles
  const int hs8  = (lane >= 16) ? 8 : 0;
  const int arow = mt * 16 + (lane & 15);
  const int col  = nt * 16 + (lane & 15);

  for (int n = 0; n < 8; ++n) {
    { // layer 1 (rank-2) fp32 -> f16 LDS
      const int c8 = t & 7;
      const float ox = obsL[bl * 16 + n * 2 + 0];
      const float oy = obsL[bl * 16 + n * 2 + 1];
#pragma unroll
      for (int i = 0; i < 16; ++i) {
        const int c = c8 * 16 + i;
        float v = fmaf(ox, W1[n * 256 + c], fmaf(oy, W1[n * 256 + 128 + c], b1[n * 128 + c]));
        h1L[bl * 136 + c] = (_Float16)(v > 0.f ? v : 0.f);
      }
    }
    __syncthreads();
    { // layer 2: [32x128]@[128x64], WMMA f16 -> f32 accum
      v8f acc = {};
#pragma unroll
      for (int k = 0; k < 4; ++k) {
        v16h a  = ldsA(&h1L[arow * 136 + k * 32 + hs8]);
        v16h bb = ldgB(w2h + (((n * 4 + nt) * 4 + k) * 512 + lane * 16));
        acc = __builtin_amdgcn_wmma_f32_16x16x32_f16(false, a, false, bb, (short)0, acc, false, false);
      }
      const float bias = b2[n * 64 + col];
#pragma unroll
      for (int r = 0; r < 8; ++r) {
        const int rr = mt * 16 + r + hs8;       // D layout: M = r (+8 for lanes>=16)
        float v = acc[r] + bias;
        hAll[rr * 520 + n * 64 + col] = (_Float16)(v > 0.f ? v : 0.f);
      }
    }
    __syncthreads();
  }

  { // pairwise Manhattan mask & invJ; pos = obs * (2, 6)
    const int i = t & 7;
    const float pix = obsL[bl * 16 + i * 2 + 0] * 2.0f;
    const float piy = obsL[bl * 16 + i * 2 + 1] * 6.0f;
    unsigned m = 0; float J = 0.f;
#pragma unroll
    for (int j = 0; j < 8; ++j) {
      const float d = fabsf(pix - obsL[bl * 16 + j * 2 + 0] * 2.0f)
                    + fabsf(piy - obsL[bl * 16 + j * 2 + 1] * 6.0f);
      const bool in = (d > 0.0f) && (d < 2.0f);
      m |= (in ? 1u : 0u) << j;
      J += in ? 1.f : 0.f;
    }
    mrowL[t] = m;
    invJL[t] = 1.0f / (J > 0.f ? J : 1.f);
  }
  __syncthreads();

  { // mixing: thread = (b, hc); each h value read once, broadcast over 8 agents
    const int hc = t & 7;
    unsigned mr[8]; float ij[8];
#pragma unroll
    for (int i = 0; i < 8; ++i) { mr[i] = mrowL[bl * 8 + i]; ij[i] = invJL[bl * 8 + i]; }
    float acc[64];
#pragma unroll
    for (int x = 0; x < 64; ++x) acc[x] = 0.f;
#pragma unroll
    for (int j = 0; j < 8; ++j) {
      v8h hv = *(const v8h*)&hAll[bl * 520 + j * 64 + hc * 8];
      float v[8];
#pragma unroll
      for (int e = 0; e < 8; ++e) v[e] = (float)hv[e];
#pragma unroll
      for (int i = 0; i < 8; ++i) {
        const float wgt = ((mr[i] >> j) & 1u) ? 1.f : 0.f;
#pragma unroll
        for (int e = 0; e < 8; ++e) acc[i * 8 + e] = fmaf(wgt, v[e], acc[i * 8 + e]);
      }
    }
#pragma unroll
    for (int i = 0; i < 8; ++i)
#pragma unroll
      for (int e = 0; e < 8; ++e) acc[i * 8 + e] *= ij[i];
    // reduce the 4 b's sharing a wave (lanes differ by 8 and 16)
#pragma unroll
    for (int x = 0; x < 64; ++x) {
      float vv = acc[x];
      vv += __shfl_xor(vv, 8, 32);
      vv += __shfl_xor(vv, 16, 32);
      acc[x] = vv;
    }
    if (lane < 8) {                      // lane == hc here
#pragma unroll
      for (int i = 0; i < 8; ++i)
#pragma unroll
        for (int e = 0; e < 8; ++e)
          atomicAdd(&commL[i * 64 + lane * 8 + e], acc[i * 8 + e]);
    }
  }
  __syncthreads();
  {
    float* dst = rep + (blockIdx.x & (NREP - 1)) * 512;
    atomicAdd(&dst[t],       commL[t]);
    atomicAdd(&dst[t + 256], commL[t + 256]);
  }
}

// ---------------- reduce replicas -> comm -> fp32 commTerm ----------------
__global__ void qnet_reduce(const float* __restrict__ rep, const float* __restrict__ bc,
                            const float* __restrict__ Wc, float* __restrict__ ct) {
  __shared__ float commL[512];
  const int t = threadIdx.x;
  float s = 0.f;
  for (int r = 0; r < NREP; ++r) s += rep[r * 512 + t];
  commL[t] = s;
  __syncthreads();
  const int n = t >> 6, c = t & 63;
  float acc = bc[n * 64 + c];
#pragma unroll 8
  for (int k = 0; k < 64; ++k)
    acc = fmaf(commL[n * 64 + k], Wc[n * 8192 + (64 + k) * 64 + c], acc);
  ct[t] = acc;
}

// ---------------- phase 2: recompute h, Wc-top GEMM + commTerm, tanh, decoder ----------------
__global__ __launch_bounds__(256) void qnet_phase2(
    const float* __restrict__ obs, const float* __restrict__ W1,
    const float* __restrict__ b1, const _Float16* __restrict__ w2h,
    const float* __restrict__ b2, const _Float16* __restrict__ wch,
    const float* __restrict__ ct, const float* __restrict__ Wd,
    const float* __restrict__ bd, float* __restrict__ qout)
{
  __shared__ float    obsL[512];
  __shared__ _Float16 h1L[32 * 136];
  __shared__ _Float16 hH[32 * 72];
  __shared__ float    h2L[32 * 68];
  __shared__ float    WdL[320];
  __shared__ float    ctL[64];
  __shared__ float    bdL[8];

  const int t    = threadIdx.x;
  const int lane = t & 31;
  const int w    = t >> 5;
  const int bl   = t >> 3;
  const int b0   = blockIdx.x * 32;

  obsL[t]       = obs[b0 * 16 + t];
  obsL[t + 256] = obs[b0 * 16 + 256 + t];
  __syncthreads();

  const int mt = w & 1, nt = w >> 1;
  const int hs8  = (lane >= 16) ? 8 : 0;
  const int arow = mt * 16 + (lane & 15);
  const int col  = nt * 16 + (lane & 15);

  for (int n = 0; n < 8; ++n) {
    if (t < 320) WdL[t] = Wd[n * 320 + t];
    if (t < 64)  ctL[t] = ct[n * 64 + t];
    if (t < 5)   bdL[t] = bd[n * 5 + t];
    { // layer 1
      const int c8 = t & 7;
      const float ox = obsL[bl * 16 + n * 2 + 0];
      const float oy = obsL[bl * 16 + n * 2 + 1];
#pragma unroll
      for (int i = 0; i < 16; ++i) {
        const int c = c8 * 16 + i;
        float v = fmaf(ox, W1[n * 256 + c], fmaf(oy, W1[n * 256 + 128 + c], b1[n * 128 + c]));
        h1L[bl * 136 + c] = (_Float16)(v > 0.f ? v : 0.f);
      }
    }
    __syncthreads();
    { // layer 2 (WMMA) -> h in f16, A-fragment-friendly row-major LDS
      v8f acc = {};
#pragma unroll
      for (int k = 0; k < 4; ++k) {
        v16h a  = ldsA(&h1L[arow * 136 + k * 32 + hs8]);
        v16h bb = ldgB(w2h + (((n * 4 + nt) * 4 + k) * 512 + lane * 16));
        acc = __builtin_amdgcn_wmma_f32_16x16x32_f16(false, a, false, bb, (short)0, acc, false, false);
      }
      const float bias = b2[n * 64 + col];
#pragma unroll
      for (int r = 0; r < 8; ++r) {
        const int rr = mt * 16 + r + hs8;
        float v = acc[r] + bias;
        hH[rr * 72 + col] = (_Float16)(v > 0.f ? v : 0.f);
      }
    }
    __syncthreads();
    { // Wc top-half GEMM (K=64) + fp32 commTerm (comm part folded exactly) -> tanh
      v8f acc = {};
#pragma unroll
      for (int k = 0; k < 2; ++k) {
        v16h a  = ldsA(&hH[arow * 72 + k * 32 + hs8]);
        v16h bb = ldgB(wch + (((n * 4 + nt) * 2 + k) * 512 + lane * 16));
        acc = __builtin_amdgcn_wmma_f32_16x16x32_f16(false, a, false, bb, (short)0, acc, false, false);
      }
      const float add = ctL[col];
#pragma unroll
      for (int r = 0; r < 8; ++r) {
        const int rr = mt * 16 + r + hs8;
        float pre = acc[r] + add;
#if __has_builtin(__builtin_amdgcn_tanhf)
        h2L[rr * 68 + col] = __builtin_amdgcn_tanhf(pre);   // v_tanh_f32 on gfx1250
#else
        h2L[rr * 68 + col] = tanhf(pre);
#endif
      }
    }
    __syncthreads();
    { // decoder 64 -> 5 (fp32 VALU)
      const int a = t & 7;
      if (a < 5) {
        float s = bdL[a];
#pragma unroll
        for (int c = 0; c < 64; ++c) s = fmaf(h2L[bl * 68 + c], WdL[c * 5 + a], s);
        qout[((long)(b0 + bl) * 8 + n) * 5 + a] = s;
      }
    }
    __syncthreads();
  }
}

extern "C" void kernel_launch(void* const* d_in, const int* in_sizes, int n_in,
                              void* d_out, int out_size, void* d_ws, size_t ws_size,
                              hipStream_t stream) {
  (void)in_sizes; (void)n_in; (void)out_size; (void)ws_size;
  const float* obs = (const float*)d_in[0];
  const float* W1  = (const float*)d_in[1];
  const float* b1  = (const float*)d_in[2];
  const float* W2  = (const float*)d_in[3];
  const float* b2  = (const float*)d_in[4];
  const float* Wc  = (const float*)d_in[5];
  const float* bc  = (const float*)d_in[6];
  const float* Wd  = (const float*)d_in[7];
  const float* bd  = (const float*)d_in[8];
  float* q = (float*)d_out;

  char* ws = (char*)d_ws;
  _Float16* w2h = (_Float16*)(ws);                            // 65536 halves (128 KB)
  _Float16* wch = (_Float16*)(ws + 131072);                   // 32768 halves (64 KB)
  float*    rep = (float*)(ws + 131072 + 65536);              // 128*512 f32 (256 KB)
  float*    ct  = (float*)(ws + 131072 + 65536 + 262144);     // 512 f32

  qnet_prep  <<<384, 256, 0, stream>>>(W2, Wc, w2h, wch);
  qnet_zero  <<<256, 256, 0, stream>>>(rep);
  qnet_phase1<<<4096, 256, 0, stream>>>(obs, W1, b1, w2h, b2, rep);
  qnet_reduce<<<1, 512, 0, stream>>>(rep, bc, Wc, ct);
  qnet_phase2<<<4096, 256, 0, stream>>>(obs, W1, b1, w2h, b2, wch, ct, Wd, bd, q);
}
